// ScaledDotProductAttention_36352603194101
// MI455X (gfx1250) — compile-verified, matrix operand order checked by reading, not run on hardware
//
#include <hip/hip_runtime.h>
#include <hip/hip_bf16.h>
#include <math.h>

// Problem constants (from reference): B=2, S=2048, H=16, D=128
#define BB 2
#define SS 2048
#define HH 16
#define DD 128

typedef _Float16 v16h __attribute__((ext_vector_type(16)));
typedef _Float16 v8h  __attribute__((ext_vector_type(8)));
typedef _Float16 v4h  __attribute__((ext_vector_type(4)));
typedef __fp16   v2hf __attribute__((ext_vector_type(2)));
typedef __fp16   v4hf __attribute__((ext_vector_type(4)));
typedef float    v8f  __attribute__((ext_vector_type(8)));

// Per-wave LDS region sizes (in halves), rows padded to kill bank conflicts.
#define K_ROW   136              // 128 + 8 pad (272B row, 16B aligned)
#define K_SZ    (32 * K_ROW)     // K chunk: [32 k][128 d]
#define VT_ROW  40               // 32 + 8 pad (80B row -> all 64 banks hit once)
#define VT_SZ   (128 * VT_ROW)   // V transposed: [128 d_v][32 k]
#define P_ROW   40               // 32 + 8 pad
#define P_SZ    (16 * P_ROW)     // P tile: [16 q][32 k]
#define WAVE_LDS (K_SZ + VT_SZ + P_SZ)   // 10112 halves = 20224 B
#define NWAVES  8

static __device__ __forceinline__ void wait_ds() {
    asm volatile("s_wait_dscnt 0" ::: "memory");
}

// Packed f32->f16 conversion: v_cvt_pk_rtz_f16_f32 x2
static __device__ __forceinline__ v4h pack4(float a, float b, float c, float d) {
    v2hf lo = __builtin_amdgcn_cvt_pkrtz(a, b);
    v2hf hi = __builtin_amdgcn_cvt_pkrtz(c, d);
    v4hf r  = __builtin_shufflevector(lo, hi, 0, 1, 2, 3);
    return __builtin_bit_cast(v4h, r);
}

// Load a 16-half WMMA fragment: two 8-half groups 16 halves apart
// (lo group = K 0..7 of this lane-half, hi group = K 16..23 of this lane-half).
static __device__ __forceinline__ v16h load_frag(const _Float16* base) {
    v8h lo = *(const v8h*)(base);
    v8h up = *(const v8h*)(base + 16);
    return __builtin_shufflevector(lo, up, 0,1,2,3,4,5,6,7,8,9,10,11,12,13,14,15);
}

__global__ __launch_bounds__(256, 2)
void flash_attn_f16_wmma(const float* __restrict__ Q,
                         const float* __restrict__ K,
                         const float* __restrict__ V,
                         const unsigned char* __restrict__ mask,
                         float* __restrict__ out) {
    const int lane = threadIdx.x & 31;
    const int wave = threadIdx.x >> 5;
    const int n    = lane & 15;   // column within 16-wide tile (C/B lane id)
    const int hif  = lane >> 4;   // half selector

    const int q0 = blockIdx.x * 16;
    const int b  = blockIdx.y >> 1;
    const int h  = (blockIdx.y & 1) * NWAVES + wave;

    __shared__ __align__(16) _Float16 smem[NWAVES * WAVE_LDS];
    _Float16* kb = &smem[wave * WAVE_LDS];   // [32][K_ROW]  K chunk, f16
    _Float16* vt = kb + K_SZ;                // [128][VT_ROW] V chunk transposed, f16
    _Float16* pb = vt + VT_SZ;               // [16][P_ROW]  P tile, f16

    const size_t rowStride = (size_t)HH * DD;   // stride between s-indices (floats)
    const float* Qb = Q + (((size_t)b * SS + q0) * HH + h) * DD;
    const float* Kb = K + ((size_t)b * SS * HH + h) * (size_t)DD;
    const float* Vb = V + ((size_t)b * SS * HH + h) * (size_t)DD;
    const unsigned char* Mb = mask + (((size_t)b * SS + q0) * SS) * HH + h;

    // ---------------- Load Q tile (16 x 128) -> f16 in kb, extract A fragments
    #pragma unroll
    for (int i = 0; i < 16; ++i) {
        const float4 qv = *(const float4*)(Qb + (size_t)i * rowStride + lane * 4);
        *(v4h*)&kb[i * K_ROW + lane * 4] = pack4(qv.x, qv.y, qv.z, qv.w);
    }
    wait_ds();
    v16h qfrag[4];
    #pragma unroll
    for (int dc = 0; dc < 4; ++dc)
        qfrag[dc] = load_frag(kb + n * K_ROW + dc * 32 + hif * 8);
    // DS is in-order per wave: later stores cannot pass these loads.

    const v8f zero8 = {0.f,0.f,0.f,0.f,0.f,0.f,0.f,0.f};
    v8f acc[8];
    #pragma unroll
    for (int vc = 0; vc < 8; ++vc) acc[vc] = zero8;
    float mrow[8], lrow[8];
    #pragma unroll
    for (int r = 0; r < 8; ++r) { mrow[r] = -3.0e38f; lrow[r] = 0.f; }

    // ---------------- Stream over keys in chunks of 32
    for (int kc = 0; kc < SS / 32; ++kc) {
        const int k0 = kc * 32;

        // Prefetch next chunk (global_prefetch_b8)
        if (kc + 1 < SS / 32) {
            const size_t nro = (size_t)(k0 + 32 + lane) * rowStride;
            __builtin_prefetch(Kb + nro, 0, 1);
            __builtin_prefetch(Vb + nro, 0, 1);
        }

        // K chunk: 32 rows x 128 d -> kb[k][d] (f16)
        #pragma unroll 4
        for (int i = 0; i < 32; ++i) {
            const float4 kv = *(const float4*)(Kb + (size_t)(k0 + i) * rowStride + lane * 4);
            *(v4h*)&kb[i * K_ROW + lane * 4] = pack4(kv.x, kv.y, kv.z, kv.w);
        }
        // V chunk transposed: vt[d_v][k], packing 4 consecutive k per b64 store
        #pragma unroll 2
        for (int g = 0; g < 8; ++g) {
            const float4 r0 = *(const float4*)(Vb + (size_t)(k0 + g*4 + 0) * rowStride + lane * 4);
            const float4 r1 = *(const float4*)(Vb + (size_t)(k0 + g*4 + 1) * rowStride + lane * 4);
            const float4 r2 = *(const float4*)(Vb + (size_t)(k0 + g*4 + 2) * rowStride + lane * 4);
            const float4 r3 = *(const float4*)(Vb + (size_t)(k0 + g*4 + 3) * rowStride + lane * 4);
            *(v4h*)&vt[(lane*4 + 0) * VT_ROW + g*4] = pack4(r0.x, r1.x, r2.x, r3.x);
            *(v4h*)&vt[(lane*4 + 1) * VT_ROW + g*4] = pack4(r0.y, r1.y, r2.y, r3.y);
            *(v4h*)&vt[(lane*4 + 2) * VT_ROW + g*4] = pack4(r0.z, r1.z, r2.z, r3.z);
            *(v4h*)&vt[(lane*4 + 3) * VT_ROW + g*4] = pack4(r0.w, r1.w, r2.w, r3.w);
        }
        wait_ds();

        // ---- Scores: S = Q (16x128) @ K^T (128x32) as two 16x16 tiles.
        // One-deep software pipeline on the B fragments so ds_load latency
        // overlaps WMMA issue (compiler can use partial s_wait_dscnt).
        v8f s0 = zero8, s1 = zero8;
        v16h bk0 = load_frag(kb + (0*16 + n) * K_ROW + hif * 8);
        v16h bk1 = load_frag(kb + (16  + n) * K_ROW + hif * 8);
        #pragma unroll
        for (int dc = 0; dc < 4; ++dc) {
            const int dn = ((dc + 1) & 3) * 32;
            v16h nb0 = load_frag(kb + (0*16 + n) * K_ROW + dn + hif * 8);
            v16h nb1 = load_frag(kb + (16  + n) * K_ROW + dn + hif * 8);
            s0 = __builtin_amdgcn_wmma_f32_16x16x32_f16(false, qfrag[dc], false, bk0,
                                                        (short)0, s0, false, false);
            s1 = __builtin_amdgcn_wmma_f32_16x16x32_f16(false, qfrag[dc], false, bk1,
                                                        (short)0, s1, false, false);
            bk0 = nb0; bk1 = nb1;
        }

        // ---- Mask + online softmax (row = r + 8*hif, spans one 16-lane half)
        #pragma unroll
        for (int r = 0; r < 8; ++r) {
            const int mq = r + 8 * hif;
            const unsigned char mb0 = Mb[((size_t)mq * SS + (k0 + n)) * HH];
            const unsigned char mb1 = Mb[((size_t)mq * SS + (k0 + 16 + n)) * HH];
            float x0 = mb0 ? 1e-9f : s0[r];
            float x1 = mb1 ? 1e-9f : s1[r];
            float t = fmaxf(x0, x1);
            t = fmaxf(t, __shfl_xor(t, 1));
            t = fmaxf(t, __shfl_xor(t, 2));
            t = fmaxf(t, __shfl_xor(t, 4));
            t = fmaxf(t, __shfl_xor(t, 8));
            const float mnew = fmaxf(mrow[r], t);
            const float sc   = __expf(mrow[r] - mnew);
            mrow[r] = mnew;
            const float p0 = __expf(x0 - mnew);
            const float p1 = __expf(x1 - mnew);
            float rs = p0 + p1;
            rs += __shfl_xor(rs, 1);
            rs += __shfl_xor(rs, 2);
            rs += __shfl_xor(rs, 4);
            rs += __shfl_xor(rs, 8);
            lrow[r] = lrow[r] * sc + rs;
            #pragma unroll
            for (int vc = 0; vc < 8; ++vc) acc[vc][r] *= sc;
            pb[mq * P_ROW + n]      = (_Float16)p0;
            pb[mq * P_ROW + 16 + n] = (_Float16)p1;
        }
        wait_ds();

        // ---- O += P (16x32) @ V (32x128), 8 v-column chunks, pipelined B loads
        const v16h pf = load_frag(pb + n * P_ROW + hif * 8);
        v16h bv = load_frag(vt + n * VT_ROW + hif * 8);
        #pragma unroll
        for (int vc = 0; vc < 8; ++vc) {
            v16h nbv = load_frag(vt + ((((vc + 1) & 7) * 16) + n) * VT_ROW + hif * 8);
            acc[vc] = __builtin_amdgcn_wmma_f32_16x16x32_f16(false, pf, false, bv,
                                                             (short)0, acc[vc], false, false);
            bv = nbv;
        }
    }

    // ---------------- Normalize and write out[b][q][h][v]
    #pragma unroll
    for (int r = 0; r < 8; ++r) {
        const int mq = r + 8 * hif;
        const float inv = 1.0f / lrow[r];
        float* op = out + (((size_t)b * SS + q0 + mq) * HH + h) * DD + n;
        #pragma unroll
        for (int vc = 0; vc < 8; ++vc)
            op[vc * 16] = acc[vc][r] * inv;
    }
}

extern "C" void kernel_launch(void* const* d_in, const int* in_sizes, int n_in,
                              void* d_out, int out_size, void* d_ws, size_t ws_size,
                              hipStream_t stream) {
    const float* Q = (const float*)d_in[0];
    const float* K = (const float*)d_in[1];
    const float* V = (const float*)d_in[2];
    const unsigned char* mask = (const unsigned char*)d_in[3];
    float* out = (float*)d_out;
    (void)in_sizes; (void)n_in; (void)out_size; (void)d_ws; (void)ws_size;

    dim3 grid(SS / 16, BB * 2, 1);   // q-tiles x (batch * head-half)
    dim3 block(32 * NWAVES, 1, 1);   // 8 waves; wave w = head (half*8 + w)
    flash_attn_f16_wmma<<<grid, block, 0, stream>>>(Q, K, V, mask, out);
}
